// FlexTwoStreamAttention_75033078662050
// MI455X (gfx1250) — matrix operand order repaired and checked by reading, not compiled, not run on hardware
//
#include <hip/hip_runtime.h>

// Problem constants (from reference): S=2048, B=1, E=1024, H=16, D=64, window=1024
#define S_LEN 2048
#define EMB   1024
#define NHEAD 16
#define HDIM  64
#define WIN   1024

typedef __bf16 bf16;
typedef __attribute__((ext_vector_type(16))) __bf16 v16bf;
typedef __attribute__((ext_vector_type(8)))  __bf16 v8bf;
typedef __attribute__((ext_vector_type(8)))  float  v8f;
typedef __attribute__((ext_vector_type(4)))  float  v4f;

static __device__ __forceinline__ v8f wmma_bf16(v16bf a, v16bf b, v8f c) {
  // D = A(16x32 bf16) x B(32x16 bf16) + C(16x16 f32)
  return __builtin_amdgcn_wmma_f32_16x16x32_bf16(
      /*neg_a=*/false, a, /*neg_b=*/false, b,
      /*c_mod=*/(short)0, c, /*reuse_a=*/false, /*reuse_b=*/false);
}

// A fragment (16x32 bf16, row-major source). rowPtr -> [row][kbase + half*8].
// Lane holds K = half*8 + 0..7 (vec[0..7]) and 16 + half*8 + 0..7 (vec[8..15]).
static __device__ __forceinline__ v16bf a_frag_bf16(const bf16* rowPtr) {
  v8bf lo = *(const v8bf*)(rowPtr);
  v8bf hi = *(const v8bf*)(rowPtr + 16);
  v16bf a;
#pragma unroll
  for (int i = 0; i < 8; ++i) { a[i] = lo[i]; a[8 + i] = hi[i]; }
  return a;
}

// ---------------------------------------------------------------------------
// One-time pack: f32 -> bf16 (8 elements per thread, b128-in / b128-out).
// ---------------------------------------------------------------------------
__global__ __launch_bounds__(256) void cvt_bf16_kernel(
    const float* __restrict__ src, bf16* __restrict__ dst, int n) {
  const int i = (blockIdx.x * 256 + threadIdx.x) * 8;
  if (i >= n) return;
  v4f a = *(const v4f*)(src + i);
  v4f b = *(const v4f*)(src + i + 4);
  v8bf o;
#pragma unroll
  for (int j = 0; j < 4; ++j) { o[j] = (bf16)a[j]; o[4 + j] = (bf16)b[j]; }
  *(v8bf*)(dst + i) = o;
}

// ---------------------------------------------------------------------------
// Projection: OUT[s][e] = (X[s][:] . Wm[e][:] + bias[e]) * scale, stored bf16.
// X and Wm are pre-packed bf16.  vlayout=0 -> dst[h][s][d];  1 -> dst[h][d][s].
// One wave computes a 16x64 output tile; K-loop over EMB in steps of 32.
// ---------------------------------------------------------------------------
__global__ __launch_bounds__(32) void proj_kernel(
    const bf16* __restrict__ X, const bf16* __restrict__ Wm,
    const float* __restrict__ bias, bf16* __restrict__ dst,
    float scale, int vlayout) {
  const int lane = threadIdx.x;
  const int r16 = lane & 15, half = lane >> 4;
  const int sbase = blockIdx.x * 16;
  const int ebase = blockIdx.y * 64;

  v8f acc[4] = {};
  const bf16* arow = X + (size_t)(sbase + r16) * EMB + half * 8;
  for (int k0 = 0; k0 < EMB; k0 += 32) {
    v16bf a = a_frag_bf16(arow + k0);
#pragma unroll
    for (int t = 0; t < 4; ++t) {
      // B frag: lane holds column n = ebase+t*16+r16, K = k0 + half*8 + 0..7 and k0 + 16 + half*8 + 0..7
      v16bf b = a_frag_bf16(Wm + (size_t)(ebase + t * 16 + r16) * EMB + k0 + half * 8);
      acc[t] = wmma_bf16(a, b, acc[t]);
    }
  }
#pragma unroll
  for (int t = 0; t < 4; ++t) {
#pragma unroll
    for (int r = 0; r < 8; ++r) {
      const int srow = sbase + r + 8 * half;
      const int e    = ebase + t * 16 + r16;
      const float v  = (acc[t][r] + bias[e]) * scale;
      const int h = e >> 6, d = e & 63;
      const size_t idx = vlayout ? (((size_t)h * HDIM + d) * S_LEN + srow)
                                 : (((size_t)h * S_LEN + srow) * HDIM + d);
      dst[idx] = (bf16)v;
    }
  }
}

// ---------------------------------------------------------------------------
// Flash attention (online softmax). One wave per (head, 16-row q tile).
// scores = Qp.K + bias  (both 1/sqrt(D) scales folded into Qp), causal+window.
// Qb,Kb: [H][S][D] bf16 row-major.  Vt: [H][D][S] bf16 (transposed).
// AO out: [S][E] bf16.
// ---------------------------------------------------------------------------
__global__ __launch_bounds__(32) void attn_kernel(
    const bf16* __restrict__ Qb, const bf16* __restrict__ Kb,
    const bf16* __restrict__ Vt, const float* __restrict__ bias,
    bf16* __restrict__ AO) {
  __shared__ bf16 pLds[16 * 32];  // P tile staging: C-layout -> A-layout

  const int lane = threadIdx.x;
  const int r16 = lane & 15, half = lane >> 4;
  const int qbase = blockIdx.x * 16;
  const int h = blockIdx.y;

  const bf16* Qh = Qb + (size_t)h * S_LEN * HDIM;
  const bf16* Kh = Kb + (size_t)h * S_LEN * HDIM;
  const bf16* Vh = Vt + (size_t)h * HDIM * S_LEN;
  const float* Bh = bias + (size_t)h * S_LEN * S_LEN;

  // Resident Q fragments: 16 rows x 64 dims = 2 A-fragments (K=32 each)
  v16bf qf[2];
#pragma unroll
  for (int c = 0; c < 2; ++c)
    qf[c] = a_frag_bf16(Qh + (size_t)(qbase + r16) * HDIM + c * 32 + half * 8);

  v8f oacc[4] = {};          // O: 16 rows x 64 dims, f32
  float m[8], lsum[8];
#pragma unroll
  for (int r = 0; r < 8; ++r) { m[r] = -3.0e38f; lsum[r] = 0.0f; }

  int kstart = qbase - WIN; if (kstart < 0) kstart = 0; kstart &= ~31;
  const int kend = qbase + 16;  // keys <= qbase+15; tail masked

  for (int kb = kstart; kb < kend; kb += 32) {
    if (kb + 32 < kend)  // stream next bias tile (global_prefetch_b8)
      __builtin_prefetch(Bh + (size_t)(qbase + 8 * half) * S_LEN + kb + 32 + r16, 0, 1);

    // S = Q (16x64) @ K^T (64x32): two 16-key N tiles, K-dim accumulated over 2 WMMAs
    v8f sacc[2] = {};
#pragma unroll
    for (int t = 0; t < 2; ++t) {
#pragma unroll
      for (int c = 0; c < 2; ++c) {
        v16bf kf = a_frag_bf16(Kh + (size_t)(kb + t * 16 + r16) * HDIM + c * 32 + half * 8);
        sacc[t] = wmma_bf16(qf[c], kf, sacc[t]);
      }
    }

    // bias + mask + online softmax (rows live in one VGPR across a 16-lane half)
    float p0v[8], p1v[8];
#pragma unroll
    for (int r = 0; r < 8; ++r) {
      const int q = qbase + r + 8 * half;
      int k0 = kb + r16;
      float s0 = sacc[0][r] + Bh[(size_t)q * S_LEN + k0];
      s0 = ((k0 <= q) && (q - k0 <= WIN)) ? s0 : -1.0e30f;
      int k1 = kb + 16 + r16;
      float s1 = sacc[1][r] + Bh[(size_t)q * S_LEN + k1];
      s1 = ((k1 <= q) && (q - k1 <= WIN)) ? s1 : -1.0e30f;

      float mx = fmaxf(s0, s1);
#pragma unroll
      for (int off = 8; off > 0; off >>= 1) mx = fmaxf(mx, __shfl_xor(mx, off, 32));
      const float mnew = fmaxf(m[r], mx);
      const float corr = __expf(m[r] - mnew);
      m[r] = mnew;
      lsum[r] *= corr;
#pragma unroll
      for (int t = 0; t < 4; ++t) oacc[t][r] = oacc[t][r] * corr;

      const float p0 = __expf(s0 - mnew);
      const float p1 = __expf(s1 - mnew);
      float ps = p0 + p1;
#pragma unroll
      for (int off = 8; off > 0; off >>= 1) ps += __shfl_xor(ps, off, 32);
      lsum[r] += ps;
      p0v[r] = p0; p1v[r] = p1;
    }

    // P: C-layout -> LDS row-major [16][32] -> A-fragment (same-wave LDS is in-order)
#pragma unroll
    for (int r = 0; r < 8; ++r) {
      pLds[(r + 8 * half) * 32 + r16]      = (bf16)p0v[r];
      pLds[(r + 8 * half) * 32 + 16 + r16] = (bf16)p1v[r];
    }
    v16bf pf = a_frag_bf16(pLds + r16 * 32 + half * 8);

    // O += P (16x32) @ V (32keys x 64dims): B from transposed V -> contiguous loads
#pragma unroll
    for (int t = 0; t < 4; ++t) {
      v16bf vf = a_frag_bf16(Vh + (size_t)(t * 16 + r16) * S_LEN + kb + half * 8);
      oacc[t] = wmma_bf16(pf, vf, oacc[t]);
    }
  }

  // normalize and store attention output (bf16, [S][E])
#pragma unroll
  for (int t = 0; t < 4; ++t) {
#pragma unroll
    for (int r = 0; r < 8; ++r) {
      const int srow = qbase + r + 8 * half;
      const int d    = t * 16 + r16;
      AO[(size_t)srow * EMB + h * HDIM + d] = (bf16)(oacc[t][r] / lsum[r]);
    }
  }
}

// ---------------------------------------------------------------------------
// Output projection: Y = AO @ Wo^T + bo; write both (identical) output streams.
// AO and Wo are bf16 (Wo pre-packed).
// ---------------------------------------------------------------------------
__global__ __launch_bounds__(32) void outproj_kernel(
    const bf16* __restrict__ AO, const bf16* __restrict__ Wo,
    const float* __restrict__ bo, float* __restrict__ out) {
  const int lane = threadIdx.x;
  const int r16 = lane & 15, half = lane >> 4;
  const int sbase = blockIdx.x * 16;
  const int ebase = blockIdx.y * 64;

  v8f acc[4] = {};
  const bf16* arow = AO + (size_t)(sbase + r16) * EMB + half * 8;
  for (int k0 = 0; k0 < EMB; k0 += 32) {
    v16bf a = a_frag_bf16(arow + k0);
#pragma unroll
    for (int t = 0; t < 4; ++t) {
      v16bf b = a_frag_bf16(Wo + (size_t)(ebase + t * 16 + r16) * EMB + k0 + half * 8);
      acc[t] = wmma_bf16(a, b, acc[t]);
    }
  }
#pragma unroll
  for (int t = 0; t < 4; ++t) {
#pragma unroll
    for (int r = 0; r < 8; ++r) {
      const int srow = sbase + r + 8 * half;
      const int e    = ebase + t * 16 + r16;
      const float v  = acc[t][r] + bo[e];
      const size_t idx = (size_t)srow * EMB + e;
      out[idx] = v;                               // c_out
      out[(size_t)S_LEN * EMB + idx] = v;         // q_out (identical stream)
    }
  }
}

extern "C" void kernel_launch(void* const* d_in, const int* in_sizes, int n_in,
                              void* d_out, int out_size, void* d_ws, size_t ws_size,
                              hipStream_t stream) {
  (void)in_sizes; (void)n_in; (void)out_size; (void)ws_size;
  const float* query = (const float*)d_in[0];
  const float* key   = (const float*)d_in[1];
  const float* value = (const float*)d_in[2];
  const float* pbias = (const float*)d_in[3];
  const float* Wq = (const float*)d_in[4];
  const float* bq = (const float*)d_in[5];
  const float* Wk = (const float*)d_in[6];
  const float* bk = (const float*)d_in[7];
  const float* Wv = (const float*)d_in[8];
  const float* bv = (const float*)d_in[9];
  const float* Wo = (const float*)d_in[10];
  const float* bo = (const float*)d_in[11];

  const size_t nSE = (size_t)S_LEN * EMB;   // 2M elements
  const size_t nEE = (size_t)EMB * EMB;     // 1M elements

  // Workspace layout (bf16 elements): Q/K/Vt heads, AO, packed X (q,k,v), packed W (q,k,v,o)
  bf16* Qb  = (bf16*)d_ws;
  bf16* Kb  = Qb  + nSE;
  bf16* Vt  = Kb  + nSE;
  bf16* AO  = Vt  + nSE;
  bf16* Xq  = AO  + nSE;
  bf16* Xk  = Xq  + nSE;
  bf16* Xv  = Xk  + nSE;
  bf16* Wqb = Xv  + nSE;
  bf16* Wkb = Wqb + nEE;
  bf16* Wvb = Wkb + nEE;
  bf16* Wob = Wvb + nEE;

  // ---- one-time f32 -> bf16 pack (activations + weights) ----
  const int gSE = (int)(nSE / 8 / 256);
  const int gEE = (int)(nEE / 8 / 256);
  cvt_bf16_kernel<<<gSE, 256, 0, stream>>>(query, Xq,  (int)nSE);
  cvt_bf16_kernel<<<gSE, 256, 0, stream>>>(key,   Xk,  (int)nSE);
  cvt_bf16_kernel<<<gSE, 256, 0, stream>>>(value, Xv,  (int)nSE);
  cvt_bf16_kernel<<<gEE, 256, 0, stream>>>(Wq,    Wqb, (int)nEE);
  cvt_bf16_kernel<<<gEE, 256, 0, stream>>>(Wk,    Wkb, (int)nEE);
  cvt_bf16_kernel<<<gEE, 256, 0, stream>>>(Wv,    Wvb, (int)nEE);
  cvt_bf16_kernel<<<gEE, 256, 0, stream>>>(Wo,    Wob, (int)nEE);

  // ---- projections (pure bf16 WMMA GEMMs) ----
  dim3 gProj(S_LEN / 16, EMB / 64);
  // Fold BOTH 1/sqrt(D) score scales into Q projection: (x@Wq^T + bq) / D
  proj_kernel<<<gProj, 32, 0, stream>>>(Xq, Wqb, bq, Qb, 1.0f / (float)HDIM, 0);
  proj_kernel<<<gProj, 32, 0, stream>>>(Xk, Wkb, bk, Kb, 1.0f, 0);
  proj_kernel<<<gProj, 32, 0, stream>>>(Xv, Wvb, bv, Vt, 1.0f, 1);  // transposed

  // ---- flash attention (bias streaming is the roofline limiter) ----
  attn_kernel<<<dim3(S_LEN / 16, NHEAD), 32, 0, stream>>>(Qb, Kb, Vt, pbias, AO);

  // ---- output projection, duplicated to both streams ----
  outproj_kernel<<<gProj, 32, 0, stream>>>(AO, Wob, bo, (float*)d_out);
}
